// LocalManifoldFFNLayer_73031623901907
// MI455X (gfx1250) — compile-verified, matrix operand order checked by reading, not run on hardware
//
#include <hip/hip_runtime.h>
#include <math.h>

// ---------- types ----------
typedef __attribute__((ext_vector_type(16))) __bf16 v16bf;
typedef __attribute__((ext_vector_type(8)))  __bf16 v8bf;
typedef __attribute__((ext_vector_type(8)))  float  v8f;

#define D_MODEL 1024
#define D_FF    4096
#define D_NEUR  128
#define N_HEADS 4
#define HEAD_D  32
#define TOPK    32
#define NTOK    8192

__device__ __forceinline__ float gelu_exact(float x) {
    return 0.5f * x * (1.0f + erff(x * 0.70710678118654752f));
}

// ---------- CDNA5 async global->LDS DMA (ASYNCcnt-tracked) ----------
__device__ __forceinline__ void async_b128(unsigned lds_addr, const __bf16* gptr) {
    asm volatile("global_load_async_to_lds_b128 %0, %1, off"
                 :: "v"(lds_addr), "v"(gptr)
                 : "memory");
}
__device__ __forceinline__ void wait_async0() {
    asm volatile("s_wait_asynccnt 0x0" ::: "memory");
}

// ---------- 1. LayerNorm -> bf16 ----------
__global__ __launch_bounds__(256)
void ln_to_bf16(const float* __restrict__ x, const float* __restrict__ g,
                const float* __restrict__ b, __bf16* __restrict__ out) {
    const int row = blockIdx.x;
    const int tid = threadIdx.x;
    const float* xr = x + (size_t)row * D_MODEL;
    __shared__ float red[256];

    float s = 0.f, s2 = 0.f;
    for (int i = tid; i < D_MODEL; i += 256) { float v = xr[i]; s += v; s2 += v * v; }

    red[tid] = s; __syncthreads();
    for (int off = 128; off > 0; off >>= 1) { if (tid < off) red[tid] += red[tid + off]; __syncthreads(); }
    const float mean = red[0] * (1.0f / D_MODEL);
    __syncthreads();
    red[tid] = s2; __syncthreads();
    for (int off = 128; off > 0; off >>= 1) { if (tid < off) red[tid] += red[tid + off]; __syncthreads(); }
    const float var  = red[0] * (1.0f / D_MODEL) - mean * mean;
    const float rstd = rsqrtf(var + 1e-5f);

    for (int i = tid; i < D_MODEL; i += 256) {
        float v = (xr[i] - mean) * rstd * g[i] + b[i];
        out[(size_t)row * D_MODEL + i] = (__bf16)v;
    }
}

// ---------- 2. f32 -> bf16 convert ----------
__global__ __launch_bounds__(256)
void f32_to_bf16(const float* __restrict__ in, __bf16* __restrict__ out, int n) {
    int i = blockIdx.x * 256 + threadIdx.x;
    if (i < n) out[i] = (__bf16)in[i];
}

// ---------- 3. bf16 WMMA GEMM: C[M,N] = act(A[M,K] @ B[N,K]^T) ----------
// Block: 128 threads (4 waves). Tile: 64(M) x 64(N) x 32(K).
// Double-buffered LDS fed by global_load_async_to_lds_b128 overlapped with WMMA.
#define BM 64
#define BN 64
#define BK 32
#define LDP 40   // BK + 8 bf16 pad (keeps 16B alignment, spreads banks)

__global__ __launch_bounds__(128)
void gemm_bf16_wmma(const __bf16* __restrict__ A, const __bf16* __restrict__ B,
                    float* __restrict__ Cf, __bf16* __restrict__ Cb,
                    int M, int N, int K, int apply_gelu) {
    __shared__ __bf16 As[2][BM][LDP];
    __shared__ __bf16 Bs[2][BN][LDP];

    const int tid  = threadIdx.x;
    const int wave = tid >> 5;
    const int lane = tid & 31;
    const int half = lane >> 4;
    const int l16  = lane & 15;

    const int mBase = blockIdx.y * BM;
    const int nBase = blockIdx.x * BN;

    v8f acc[4] = {};

    // tile copier mapping: each thread DMAs 2x16B for A and 2x16B for B
    const int lrow = tid >> 2;         // 0..31
    const int lcol = (tid & 3) * 8;    // 0,8,16,24 (bf16 units)

    const __bf16* aG0 = A + (size_t)(mBase + lrow)      * K + lcol;
    const __bf16* aG1 = A + (size_t)(mBase + lrow + 32) * K + lcol;
    const __bf16* bG0 = B + (size_t)(nBase + lrow)      * K + lcol;
    const __bf16* bG1 = B + (size_t)(nBase + lrow + 32) * K + lcol;

    unsigned ldsA0[2], ldsA1[2], ldsB0[2], ldsB1[2];
    #pragma unroll
    for (int p = 0; p < 2; p++) {
        ldsA0[p] = (unsigned)(uintptr_t)&As[p][lrow][lcol];
        ldsA1[p] = (unsigned)(uintptr_t)&As[p][lrow + 32][lcol];
        ldsB0[p] = (unsigned)(uintptr_t)&Bs[p][lrow][lcol];
        ldsB1[p] = (unsigned)(uintptr_t)&Bs[p][lrow + 32][lcol];
    }

    // prologue: DMA tile 0 into buffer 0
    async_b128(ldsA0[0], aG0);
    async_b128(ldsA1[0], aG1);
    async_b128(ldsB0[0], bG0);
    async_b128(ldsB1[0], bG1);

    const int T = K / BK;
    for (int t = 0; t < T; t++) {
        wait_async0();        // this wave's DMA for tile t has landed in LDS
        __syncthreads();      // all waves' DMAs landed; all waves done reading prev buf

        if (t + 1 < T) {      // overlap: DMA tile t+1 while computing tile t
            const int kn = (t + 1) * BK;
            const int pn = (t + 1) & 1;
            async_b128(ldsA0[pn], aG0 + kn);
            async_b128(ldsA1[pn], aG1 + kn);
            async_b128(ldsB0[pn], bG0 + kn);
            async_b128(ldsB1[pn], bG1 + kn);
        }

        const int p = t & 1;
        // A fragment: lane holds row (wave*16 + l16); lanes16-31 take K halves +8/+24
        const int ar = wave * 16 + l16;
        v8bf alo = *(const v8bf*)(&As[p][ar][half * 8]);
        v8bf ahi = *(const v8bf*)(&As[p][ar][16 + half * 8]);
        v16bf afrag = __builtin_shufflevector(alo, ahi,
            0,1,2,3,4,5,6,7,8,9,10,11,12,13,14,15);

        #pragma unroll
        for (int nb = 0; nb < 4; nb++) {
            const int br = nb * 16 + l16;     // B^T row == W output feature
            v8bf blo = *(const v8bf*)(&Bs[p][br][half * 8]);
            v8bf bhi = *(const v8bf*)(&Bs[p][br][16 + half * 8]);
            v16bf bfrag = __builtin_shufflevector(blo, bhi,
                0,1,2,3,4,5,6,7,8,9,10,11,12,13,14,15);
            acc[nb] = __builtin_amdgcn_wmma_f32_16x16x32_bf16(
                false, afrag, false, bfrag, (short)0, acc[nb], false, false);
        }
    }

    // C/D layout: VGPR r -> (lanes 0-15: M=r, N=lane) (lanes 16-31: M=8+r, N=lane-16)
    const int mRow0 = mBase + wave * 16 + half * 8;
    #pragma unroll
    for (int nb = 0; nb < 4; nb++) {
        const int n = nBase + nb * 16 + l16;
        #pragma unroll
        for (int r = 0; r < 8; r++) {
            const int m = mRow0 + r;
            float v = acc[nb][r];
            if (apply_gelu) v = gelu_exact(v);
            if (Cf) Cf[(size_t)m * N + n] = v;
            if (Cb) Cb[(size_t)m * N + n] = (__bf16)v;
        }
    }
}

// ---------- 4. top-32 per token over 4096 scores ----------
__global__ __launch_bounds__(256)
void topk32(const float* __restrict__ scores, int* __restrict__ idx) {
    const int row = blockIdx.x;
    const int tid = threadIdx.x;
    __shared__ float s[D_FF];
    __shared__ float rv[256];
    __shared__ int   ri[256];

    const float* sr = scores + (size_t)row * D_FF;
    for (int i = tid; i < D_FF; i += 256) s[i] = sr[i];
    __syncthreads();

    for (int it = 0; it < TOPK; it++) {
        float bv = -INFINITY; int bi = 0x7fffffff;
        for (int i = tid; i < D_FF; i += 256) {
            float v = s[i];
            if (v > bv) { bv = v; bi = i; }
        }
        rv[tid] = bv; ri[tid] = bi;
        __syncthreads();
        for (int off = 128; off > 0; off >>= 1) {
            if (tid < off) {
                float ov = rv[tid + off]; int oi = ri[tid + off];
                if (ov > rv[tid] || (ov == rv[tid] && oi < ri[tid])) { rv[tid] = ov; ri[tid] = oi; }
            }
            __syncthreads();
        }
        if (tid == 0) { idx[row * TOPK + it] = ri[0]; s[ri[0]] = -INFINITY; }
        __syncthreads();
    }
}

// ---------- 5. per-token neuron attention + weight-projection MLP ----------
// one block (128 thr) per token; dynamic LDS (~96 KB, fine on 320 KB/WGP)
__global__ __launch_bounds__(128)
void neuron_attn(const int* __restrict__ idx, const float* __restrict__ nvecs,
                 const float* __restrict__ in_w, const float* __restrict__ in_b,
                 const float* __restrict__ out_w, const float* __restrict__ out_b,
                 const float* __restrict__ wp1_w, const float* __restrict__ wp1_b,
                 const float* __restrict__ wp2_w, const float* __restrict__ wp2_b,
                 float* __restrict__ wgt) {
    extern __shared__ float sm[];
    float* infos = sm;            // 32*128
    float* qkv   = sm + 4096;     // 32*384
    float* attn  = sm + 16384;    // 4*32*32
    float* ctx   = sm + 20480;    // 32*128
    float* attended = infos;      // reuse (infos dead after qkv)
    float* hidden   = attn;       // reuse (attn dead after ctx) 32*64

    const int t   = blockIdx.x;
    const int tid = threadIdx.x;
    const int* idr = idx + t * TOPK;

    for (int i = tid; i < TOPK * D_NEUR; i += 128) {
        int kk = i >> 7, d = i & 127;
        infos[i] = nvecs[(size_t)idr[kk] * D_NEUR + d];
    }
    __syncthreads();

    // qkv[32][384] = infos @ in_w^T + in_b
    for (int i = tid; i < TOPK * 384; i += 128) {
        int kk = i / 384, e = i - kk * 384;
        const float* wr = in_w + (size_t)e * D_NEUR;
        const float* xr = infos + kk * D_NEUR;
        float a = in_b[e];
        #pragma unroll 8
        for (int d = 0; d < D_NEUR; d++) a = fmaf(xr[d], wr[d], a);
        qkv[i] = a;
    }
    __syncthreads();

    // attention scores per head: s[h][q][k]
    const float scale = 0.17677669529663688f;  // 1/sqrt(32)
    for (int i = tid; i < N_HEADS * TOPK * TOPK; i += 128) {
        int h = i >> 10, rem = i & 1023, q = rem >> 5, k = rem & 31;
        const float* qp = qkv + q * 384 + h * HEAD_D;
        const float* kp = qkv + k * 384 + D_NEUR + h * HEAD_D;
        float a = 0.f;
        #pragma unroll
        for (int d = 0; d < HEAD_D; d++) a = fmaf(qp[d], kp[d], a);
        attn[i] = a * scale;
    }
    __syncthreads();

    // softmax over k: 128 rows, one per thread
    {
        float* rp = attn + (tid >> 5) * 1024 + (tid & 31) * 32;
        float mx = rp[0];
        for (int k = 1; k < TOPK; k++) mx = fmaxf(mx, rp[k]);
        float s = 0.f;
        for (int k = 0; k < TOPK; k++) { float e = __expf(rp[k] - mx); rp[k] = e; s += e; }
        float inv = 1.0f / s;
        for (int k = 0; k < TOPK; k++) rp[k] *= inv;
    }
    __syncthreads();

    // ctx[q][h*32+d] = sum_k p[h][q][k] * v[k][h*32+d]
    for (int i = tid; i < TOPK * D_NEUR; i += 128) {
        int q = i >> 7, e = i & 127, h = e >> 5, d = e & 31;
        const float* pr = attn + h * 1024 + q * 32;
        float a = 0.f;
        #pragma unroll
        for (int k = 0; k < TOPK; k++) a = fmaf(pr[k], qkv[k * 384 + 256 + h * HEAD_D + d], a);
        ctx[i] = a;
    }
    __syncthreads();

    // attended = ctx @ out_w^T + out_b
    for (int i = tid; i < TOPK * D_NEUR; i += 128) {
        int kk = i >> 7, e = i & 127;
        const float* wr = out_w + (size_t)e * D_NEUR;
        const float* xr = ctx + kk * D_NEUR;
        float a = out_b[e];
        #pragma unroll 8
        for (int d = 0; d < D_NEUR; d++) a = fmaf(xr[d], wr[d], a);
        attended[i] = a;
    }
    __syncthreads();

    // hidden = gelu(attended @ wp1^T + b1)   [32][64]
    for (int i = tid; i < TOPK * 64; i += 128) {
        int kk = i >> 6, j = i & 63;
        const float* wr = wp1_w + (size_t)j * D_NEUR;
        const float* xr = attended + kk * D_NEUR;
        float a = wp1_b[j];
        #pragma unroll 8
        for (int d = 0; d < D_NEUR; d++) a = fmaf(xr[d], wr[d], a);
        hidden[i] = gelu_exact(a);
    }
    __syncthreads();

    // w[kk] = hidden @ wp2^T + b2
    if (tid < TOPK) {
        const float* xr = hidden + tid * 64;
        float a = wp2_b[0];
        #pragma unroll 8
        for (int d = 0; d < 64; d++) a = fmaf(xr[d], wp2_w[d], a);
        wgt[(size_t)t * TOPK + tid] = a;
    }
}

// ---------- 6. out[t][d] = sum_k w[t][k] * dirs[idx[t][k]][d] ----------
__global__ __launch_bounds__(256)
void weighted_dirs(const int* __restrict__ idx, const float* __restrict__ wgt,
                   const float* __restrict__ dirs, float* __restrict__ out) {
    const int t = blockIdx.x;
    __shared__ int   sid[TOPK];
    __shared__ float sw[TOPK];
    if (threadIdx.x < TOPK) {
        sid[threadIdx.x] = idx[t * TOPK + threadIdx.x];
        sw[threadIdx.x]  = wgt[t * TOPK + threadIdx.x];
    }
    __syncthreads();
    for (int d = threadIdx.x; d < D_MODEL; d += 256) {
        float a = 0.f;
        #pragma unroll 8
        for (int k = 0; k < TOPK; k++)
            a = fmaf(sw[k], dirs[(size_t)sid[k] * D_MODEL + d], a);
        out[(size_t)t * D_MODEL + d] = a;
    }
}

// ---------- launcher ----------
extern "C" void kernel_launch(void* const* d_in, const int* in_sizes, int n_in,
                              void* d_out, int out_size, void* d_ws, size_t ws_size,
                              hipStream_t stream) {
    const float* x      = (const float*)d_in[0];
    const float* W1     = (const float*)d_in[1];
    const float* W2     = (const float*)d_in[2];
    const float* ln_g   = (const float*)d_in[3];
    const float* ln_b   = (const float*)d_in[4];
    const float* nvecs  = (const float*)d_in[5];
    const float* ndirs  = (const float*)d_in[6];
    const float* in_w   = (const float*)d_in[7];
    const float* in_b   = (const float*)d_in[8];
    const float* out_w  = (const float*)d_in[9];
    const float* out_b  = (const float*)d_in[10];
    const float* wp1_w  = (const float*)d_in[11];
    const float* wp1_b  = (const float*)d_in[12];
    const float* wp2_w  = (const float*)d_in[13];
    const float* wp2_b  = (const float*)d_in[14];
    float* out = (float*)d_out;

    // workspace carve (256B aligned)
    char* ws = (char*)d_ws;
    size_t off = 0;
    auto carve = [&](size_t bytes) { char* p = ws + off; off += (bytes + 255) & ~(size_t)255; return p; };
    __bf16* xnb    = (__bf16*)carve((size_t)NTOK * D_MODEL * 2);
    __bf16* w1b    = (__bf16*)carve((size_t)D_MODEL * D_MODEL * 2);
    __bf16* w2b    = (__bf16*)carve((size_t)D_FF * D_MODEL * 2);
    __bf16* hb     = (__bf16*)carve((size_t)NTOK * D_MODEL * 2);
    float*  scores = (float*)carve((size_t)NTOK * D_FF * 4);
    int*    idxw   = (int*)carve((size_t)NTOK * TOPK * 4);
    float*  wgtw   = (float*)carve((size_t)NTOK * TOPK * 4);

    // 1. LN -> bf16
    ln_to_bf16<<<NTOK, 256, 0, stream>>>(x, ln_g, ln_b, xnb);
    // 2. weight converts
    f32_to_bf16<<<(D_MODEL * D_MODEL + 255) / 256, 256, 0, stream>>>(W1, w1b, D_MODEL * D_MODEL);
    f32_to_bf16<<<(D_FF * D_MODEL + 255) / 256, 256, 0, stream>>>(W2, w2b, D_FF * D_MODEL);
    // 3. GEMM1: h = gelu(xn @ W1^T) -> bf16
    gemm_bf16_wmma<<<dim3(D_MODEL / BN, NTOK / BM), 128, 0, stream>>>(
        xnb, w1b, nullptr, hb, NTOK, D_MODEL, D_MODEL, 1);
    // 4. GEMM2: scores = h @ W2^T -> f32
    gemm_bf16_wmma<<<dim3(D_FF / BN, NTOK / BM), 128, 0, stream>>>(
        hb, w2b, scores, nullptr, NTOK, D_FF, D_MODEL, 0);
    // 5. top-32
    topk32<<<NTOK, 256, 0, stream>>>(scores, idxw);
    // 6. per-token attention + weight MLP
    const size_t smem = (size_t)(4096 + 12288 + 4096 + 4096) * sizeof(float);
    neuron_attn<<<NTOK, 128, smem, stream>>>(idxw, nvecs, in_w, in_b, out_w, out_b,
                                             wp1_w, wp1_b, wp2_w, wp2_b, wgtw);
    // 7. weighted direction sum -> output
    weighted_dirs<<<NTOK, 256, 0, stream>>>(idxw, wgtw, ndirs, out);
    (void)in_sizes; (void)n_in; (void)out_size; (void)ws_size;
}
